// AnatomicalTextEnhancer_57964878626838
// MI455X (gfx1250) — compile-verified, matrix operand order checked by reading, not run on hardware
//
#include <hip/hip_runtime.h>
#include <hip/hip_bf16.h>

typedef __attribute__((ext_vector_type(16))) _Float16 v16h;
typedef __attribute__((ext_vector_type(8)))  _Float16 half8;
typedef __attribute__((ext_vector_type(8)))  float    v8f;

#define B_ 16
#define R_ 29
#define D_ 512
#define N_ 20000
#define TOPK 5
#define MASK_VAL  (-1.0e9f)
#define SENT_VAL  (-3.0e38f)

#define WAVES_PER_BLK 4
#define WAVE_SPAN 256                        // n-columns per wave (8 groups of 32)
#define BLKS_PER_REGION 20                   // ceil(20000 / (4*256)) = 20
#define PSLOTS (BLKS_PER_REGION * WAVES_PER_BLK)   // 80 partial slots per region
#define LDS_PITCH 520                        // 512 + 8 halves: kills bank conflicts

#define SQ4(v) ((v).x*(v).x + (v).y*(v).y + (v).z*(v).z + (v).w*(v).w)

// ---------------------------------------------------------------------------
// Kernel 1: L2-normalize queries (eps like F.normalize) -> f16, layout [r][b][d]
// One wave per (b, r) pair; lane handles 16 contiguous d's.
// ---------------------------------------------------------------------------
__global__ void qnorm_kernel(const float* __restrict__ q, _Float16* __restrict__ qh) {
    int w    = (blockIdx.x * blockDim.x + threadIdx.x) >> 5;
    int lane = threadIdx.x & 31;
    if (w >= B_ * R_) return;
    int b = w / R_, r = w % R_;

    const float* src = q + ((size_t)b * R_ + r) * D_ + lane * 16;
    float4 x0 = ((const float4*)src)[0];
    float4 x1 = ((const float4*)src)[1];
    float4 x2 = ((const float4*)src)[2];
    float4 x3 = ((const float4*)src)[3];

    float ss = SQ4(x0) + SQ4(x1) + SQ4(x2) + SQ4(x3);
    #pragma unroll
    for (int m = 1; m < 32; m <<= 1) ss += __shfl_xor(ss, m);
    float sc = 1.0f / fmaxf(sqrtf(ss), 1e-12f);

    union { _Float16 h[16]; uint4 u[2]; } pk;
    float f[16] = { x0.x,x0.y,x0.z,x0.w, x1.x,x1.y,x1.z,x1.w,
                    x2.x,x2.y,x2.z,x2.w, x3.x,x3.y,x3.z,x3.w };
    #pragma unroll
    for (int i = 0; i < 16; ++i) pk.h[i] = (_Float16)(f[i] * sc);

    _Float16* dst = qh + ((size_t)r * B_ + b) * D_ + lane * 16;
    ((uint4*)dst)[0] = pk.u[0];
    ((uint4*)dst)[1] = pk.u[1];
}

// convert 4 f32x4 chunks into one 16-half fragment (WMMA 16-bit K layout)
__device__ __forceinline__ v16h cvt_frag(float4 x0, float4 x1, float4 x2, float4 x3) {
    v16h bb;
    bb[0]=(_Float16)x0.x; bb[1]=(_Float16)x0.y; bb[2]=(_Float16)x0.z; bb[3]=(_Float16)x0.w;
    bb[4]=(_Float16)x1.x; bb[5]=(_Float16)x1.y; bb[6]=(_Float16)x1.z; bb[7]=(_Float16)x1.w;
    bb[8]=(_Float16)x2.x; bb[9]=(_Float16)x2.y; bb[10]=(_Float16)x2.z; bb[11]=(_Float16)x2.w;
    bb[12]=(_Float16)x3.x; bb[13]=(_Float16)x3.y; bb[14]=(_Float16)x3.z; bb[15]=(_Float16)x3.w;
    return bb;
}

// ---------------------------------------------------------------------------
// Kernel 2: per-region WMMA GEMM (16 x Nspan) + fused kb-row norm + mask +
// running register top-5, partials to workspace.
// grid.x = R_*BLKS_PER_REGION, block = 128 (4 waves).
// Each wave iteration handles TWO 16-column tiles sharing one A fragment:
// 8x global_load_b128 per k-step in flight -> deep HBM pipelining.
// ---------------------------------------------------------------------------
__global__ __launch_bounds__(128)
void sim_topk_kernel(const _Float16* __restrict__ qh,
                     const float*    __restrict__ kb,
                     const int*      __restrict__ qids_g,
                     const int*      __restrict__ kbids_g,
                     float* __restrict__ pv, int* __restrict__ pi) {
    __shared__ __attribute__((aligned(16))) _Float16 qlds[B_ * LDS_PITCH];
    __shared__ float s_mv[WAVES_PER_BLK][B_][16 * TOPK];
    __shared__ int   s_mi[WAVES_PER_BLK][B_][16 * TOPK];

    int r    = blockIdx.x / BLKS_PER_REGION;
    int blk  = blockIdx.x % BLKS_PER_REGION;
    int tid  = threadIdx.x;
    int wave = tid >> 5;
    int lane = tid & 31;

    // Stage this region's normalized Q (16x512 f16) into padded LDS.
    for (int c = tid; c < B_ * (D_ / 8); c += blockDim.x) {
        int row = c >> 6, col8 = c & 63;
        uint4 v = *(const uint4*)(qh + ((size_t)r * B_ + row) * D_ + col8 * 8);
        *(uint4*)&qlds[row * LDS_PITCH + col8 * 8] = v;
    }
    __syncthreads();

    int half  = lane >> 4;      // which K-half this lane feeds (WMMA 16-bit layout)
    int ln    = lane & 15;      // matrix row (A: M) / column (B: N) index
    int khalf = half * 8;

    int qids[8];
    #pragma unroll
    for (int j = 0; j < 8; ++j) qids[j] = qids_g[half * 8 + j];

    float tv[8][TOPK]; int ti[8][TOPK];
    #pragma unroll
    for (int j = 0; j < 8; ++j)
        #pragma unroll
        for (int s = 0; s < TOPK; ++s) { tv[j][s] = SENT_VAL; ti[j][s] = 0; }

    int n0 = (blk * WAVES_PER_BLK + wave) * WAVE_SPAN;
    int n1 = n0 + WAVE_SPAN; if (n1 > N_) n1 = N_;

    for (int nb = n0; nb < n1; nb += 32) {            // two 16-wide tiles / group
        int na = nb + ln, nc = nb + 16 + ln;
        const float* rowa = kb + ((size_t)r * N_ + na) * D_ + khalf;
        const float* rowc = kb + ((size_t)r * N_ + nc) * D_ + khalf;
        __builtin_prefetch(rowa + 32 * D_, 0, 1);     // next group, same lane-rows
        __builtin_prefetch(rowc + 32 * D_, 0, 1);

        v8f c0 = {}, c1 = {};
        float ssa = 0.0f, ssc = 0.0f;
        #pragma unroll 2
        for (int k0 = 0; k0 < D_; k0 += 32) {
            float4 a0 = *(const float4*)(rowa + k0);
            float4 a1 = *(const float4*)(rowa + k0 + 4);
            float4 a2 = *(const float4*)(rowa + k0 + 16);
            float4 a3 = *(const float4*)(rowa + k0 + 20);
            float4 b0 = *(const float4*)(rowc + k0);
            float4 b1 = *(const float4*)(rowc + k0 + 4);
            float4 b2 = *(const float4*)(rowc + k0 + 16);
            float4 b3 = *(const float4*)(rowc + k0 + 20);
            ssa += SQ4(a0) + SQ4(a1) + SQ4(a2) + SQ4(a3);
            ssc += SQ4(b0) + SQ4(b1) + SQ4(b2) + SQ4(b3);

            v16h fa = cvt_frag(a0, a1, a2, a3);
            v16h fc = cvt_frag(b0, b1, b2, b3);

            half8 q0 = *(const half8*)&qlds[ln * LDS_PITCH + k0 + khalf];
            half8 q1 = *(const half8*)&qlds[ln * LDS_PITCH + k0 + 16 + khalf];
            v16h aa;
            #pragma unroll
            for (int e = 0; e < 8; ++e) { aa[e] = q0[e]; aa[8 + e] = q1[e]; }

            c0 = __builtin_amdgcn_wmma_f32_16x16x32_f16(
                     false, aa, false, fa, (short)0, c0, false, false);
            c1 = __builtin_amdgcn_wmma_f32_16x16x32_f16(
                     false, aa, false, fc, (short)0, c1, false, false);
        }

        // finish kb-row norms: lanes l and l^16 hold disjoint halves of the row
        float ta = ssa + __shfl_xor(ssa, 16);
        float tc = ssc + __shfl_xor(ssc, 16);
        float sca = 1.0f / fmaxf(sqrtf(ta), 1e-12f);
        float scc = 1.0f / fmaxf(sqrtf(tc), 1e-12f);
        int ida = kbids_g[(size_t)r * N_ + na];
        int idc = kbids_g[(size_t)r * N_ + nc];

        #pragma unroll
        for (int j = 0; j < 8; ++j) {
            float v0 = (ida == qids[j]) ? MASK_VAL : c0[j] * sca;
            float v1 = (idc == qids[j]) ? MASK_VAL : c1[j] * scc;
            float v = v0; int ii = na;
            #pragma unroll
            for (int s = 0; s < TOPK; ++s) {          // branchless sorted insert
                bool sw = v > tv[j][s];
                float ov = tv[j][s]; int oi = ti[j][s];
                tv[j][s] = sw ? v : ov;  ti[j][s] = sw ? ii : oi;
                v = sw ? ov : v;  ii = sw ? oi : ii;
            }
            v = v1; ii = nc;
            #pragma unroll
            for (int s = 0; s < TOPK; ++s) {
                bool sw = v > tv[j][s];
                float ov = tv[j][s]; int oi = ti[j][s];
                tv[j][s] = sw ? v : ov;  ti[j][s] = sw ? ii : oi;
                v = sw ? ov : v;  ii = sw ? oi : ii;
            }
        }
    }

    // dump per-lane top-5 lists, merge 16 lanes -> one top-5 per (b, wave)
    #pragma unroll
    for (int j = 0; j < 8; ++j) {
        int b = half * 8 + j;
        #pragma unroll
        for (int s = 0; s < TOPK; ++s) {
            s_mv[wave][b][ln * TOPK + s] = tv[j][s];
            s_mi[wave][b][ln * TOPK + s] = ti[j][s];
        }
    }
    __syncthreads();

    if (lane < B_) {
        int b = lane;
        float mv[TOPK]; int mi[TOPK];
        #pragma unroll
        for (int s = 0; s < TOPK; ++s) { mv[s] = SENT_VAL; mi[s] = 0; }
        for (int e = 0; e < 16 * TOPK; ++e) {
            float v = s_mv[wave][b][e]; int ii = s_mi[wave][b][e];
            #pragma unroll
            for (int s = 0; s < TOPK; ++s) {
                bool sw = v > mv[s];
                float ov = mv[s]; int oi = mi[s];
                mv[s] = sw ? v : ov;  mi[s] = sw ? ii : oi;
                v = sw ? ov : v;  ii = sw ? oi : ii;
            }
        }
        int slot = blk * WAVES_PER_BLK + wave;
        size_t base = (((size_t)r * PSLOTS + slot) * B_ + b) * TOPK;
        #pragma unroll
        for (int s = 0; s < TOPK; ++s) { pv[base + s] = mv[s]; pi[base + s] = mi[s]; }
    }
}

// ---------------------------------------------------------------------------
// Kernel 3: merge 80 partials per (b,r) -> similarity_scores, top_vals, top_idx
// ---------------------------------------------------------------------------
__global__ void final_topk_kernel(const float* __restrict__ pv,
                                  const int*   __restrict__ pi,
                                  float* __restrict__ out) {
    int t = blockIdx.x * blockDim.x + threadIdx.x;
    if (t >= B_ * R_) return;
    int b = t / R_, r = t % R_;          // t == b*R_ + r (row-major [B,R])

    float mv[TOPK]; int mi[TOPK];
    #pragma unroll
    for (int s = 0; s < TOPK; ++s) { mv[s] = SENT_VAL; mi[s] = 0; }

    for (int slot = 0; slot < PSLOTS; ++slot) {
        size_t base = (((size_t)r * PSLOTS + slot) * B_ + b) * TOPK;
        #pragma unroll
        for (int e = 0; e < TOPK; ++e) {
            float v = pv[base + e]; int ii = pi[base + e];
            #pragma unroll
            for (int s = 0; s < TOPK; ++s) {
                bool sw = v > mv[s];
                float ov = mv[s]; int oi = mi[s];
                mv[s] = sw ? v : ov;  mi[s] = sw ? ii : oi;
                v = sw ? ov : v;  ii = sw ? oi : ii;
            }
        }
    }

    out[t] = mv[0];                                    // similarity_scores [B,R]
    float* otv = out + B_ * R_;                        // top_vals [B,R,5]
    float* oti = out + B_ * R_ + B_ * R_ * TOPK;       // top_idx  [B,R,5]
    #pragma unroll
    for (int s = 0; s < TOPK; ++s) {
        otv[(size_t)t * TOPK + s] = mv[s];
        oti[(size_t)t * TOPK + s] = (float)mi[s];
    }
}

// ---------------------------------------------------------------------------
extern "C" void kernel_launch(void* const* d_in, const int* in_sizes, int n_in,
                              void* d_out, int out_size, void* d_ws, size_t ws_size,
                              hipStream_t stream) {
    const float* q     = (const float*)d_in[0];   // [16,29,512]
    const float* kb    = (const float*)d_in[1];   // [29,20000,512]
    const int*   qids  = (const int*)d_in[2];     // [16]
    const int*   kbids = (const int*)d_in[3];     // [29,20000]
    (void)in_sizes; (void)n_in; (void)out_size; (void)ws_size; // top_k fixed = 5

    // workspace layout (~1.96 MB total):
    char* ws = (char*)d_ws;
    _Float16* qh = (_Float16*)ws;                                 // 475,136 B
    size_t qh_bytes = (size_t)R_ * B_ * D_ * sizeof(_Float16);
    float* pv = (float*)(ws + qh_bytes);                          // 742,400 B
    size_t pv_bytes = (size_t)R_ * PSLOTS * B_ * TOPK * sizeof(float);
    int*   pi = (int*)(ws + qh_bytes + pv_bytes);                 // 742,400 B

    float* out = (float*)d_out;   // 464 + 2320 + 2320 = 5104 floats

    qnorm_kernel<<<(B_ * R_ * 32 + 255) / 256, 256, 0, stream>>>(q, qh);
    sim_topk_kernel<<<R_ * BLKS_PER_REGION, 128, 0, stream>>>(qh, kb, qids, kbids, pv, pi);
    final_topk_kernel<<<(B_ * R_ + 127) / 128, 128, 0, stream>>>(pv, pi, out);
}
